// Model_78572131713308
// MI455X (gfx1250) — matrix operand order repaired and checked by reading, not compiled, not run on hardware
//
#include <hip/hip_runtime.h>

// CDNA5 / gfx1250, wave32. Spiral-mesh autoencoder.
// Whole intermediate pipeline held in f16 (the WMMA input precision), halving
// the dominant gather/stream bandwidth. Spiral convs: fused gather +
// v_wmma_f32_16x16x32_f16 with fragments loaded as raw v8h vectors (no LDS,
// no per-element converts). Pools: scatter-free K=3 gathers. Latent linears:
// pure bandwidth kernels (M=8 -> WMMA useless there).

typedef __attribute__((ext_vector_type(16))) _Float16 v16h;
typedef __attribute__((ext_vector_type(8)))  _Float16 v8h;
typedef __attribute__((ext_vector_type(8)))  float    v8f;

// ---------------------------------------------------------------------------
// f32 -> f16 conversion (inputs/weights), run once per launch; ~2.4M elems tot.
// ---------------------------------------------------------------------------
__global__ void cvt_f32_f16(const float* __restrict__ src, _Float16* __restrict__ dst, int n)
{
  int i = blockIdx.x * blockDim.x + threadIdx.x;
  if (i < n) dst[i] = (_Float16)src[i];
}

// ---------------------------------------------------------------------------
// Fused spiral gather + GEMM. One wave = one 16-node tile x all co-tiles.
// A lane holds node row m=lane&15; per K-chunk its elements are two contiguous
// 8-f16 runs -> two 16B loads. B lane holds W row co; same two 8-f16 runs ->
// two 16B loads. Fragment pack = shufflevector. (ISA 7.12.2 wave32 layouts.)
// ---------------------------------------------------------------------------
template<int CIN, int COUT, typename TOUT>
__global__ __launch_bounds__(256) void spiral_conv_wmma(
    const _Float16* __restrict__ in, const int* __restrict__ spiral,
    const _Float16* __restrict__ W, const float* __restrict__ bias,
    TOUT* __restrict__ out, int nNodes, int applyElu)
{
  constexpr int KTOT    = 9 * CIN;
  constexpr int COTILES = (COUT + 15) / 16;
  constexpr int KCHUNKS = (KTOT + 31) / 32;

  const int lane = threadIdx.x & 31;
  const int wave = threadIdx.x >> 5;
  const int half = lane >> 4;        // K/M lane-group split per ISA 7.12.2
  const int m    = lane & 15;        // A row during load, D column at store

  const int nodeBase = (blockIdx.x * 8 + wave) * 16;
  if (nodeBase >= nNodes) return;
  const int b = blockIdx.y;

  const int nodeRaw = nodeBase + m;
  const int node    = (nodeRaw < nNodes) ? nodeRaw : (nNodes - 1); // clamp, masked at store
  const _Float16* inb = in + (size_t)b * nNodes * CIN;
  const int*      spr = spiral + (size_t)node * 9;

  v8f acc[COTILES];
  #pragma unroll
  for (int t = 0; t < COTILES; ++t) acc[t] = (v8f){};

  for (int kc = 0; kc < KCHUNKS; ++kc) {
    // ---------------- A fragment (gathered activations) ----------------
    v16h aF;
    if constexpr (CIN >= 32 && (KTOT % 32) == 0) {
      constexpr int CPL = CIN / 32;                 // K-chunks per spiral slot
      const int l   = kc / CPL;
      const int ci0 = (kc - l * CPL) * 32 + half * 8;
      const _Float16* src = inb + (size_t)spr[l] * CIN + ci0;
      const v8h lo = *(const v8h*)(src);            // k = ci0 + 0..7
      const v8h hi = *(const v8h*)(src + 16);       // k = ci0 + 16..23
      aF = __builtin_shufflevector(lo, hi, 0, 1, 2, 3, 4, 5, 6, 7,
                                           8, 9, 10, 11, 12, 13, 14, 15);
    } else {
      // generic guarded path (CIN=3 head: KTOT=27, single chunk)
      #pragma unroll
      for (int j = 0; j < 16; ++j) {
        const int p = j >> 1, h = j & 1;
        const int k = kc * 32 + ((p & 4) ? 16 : 0) + half * 8 + (p & 3) * 2 + h;
        _Float16 v = (_Float16)0.f;
        if (k < KTOT) {
          const int l  = k / CIN;
          const int ci = k - l * CIN;
          v = inb[(size_t)spr[l] * CIN + ci];
        }
        aF[j] = v;
      }
    }

    // ---------------- B fragments + WMMA per co-tile ----------------
    #pragma unroll
    for (int t = 0; t < COTILES; ++t) {
      const int co = t * 16 + m;
      v16h bF;
      if constexpr ((KTOT % 32) == 0 && (COUT % 16) == 0) {
        const _Float16* wrow = W + (size_t)co * KTOT + kc * 32 + half * 8;
        if (kc + 1 < KCHUNKS) __builtin_prefetch(wrow + 32, 0, 1); // global_prefetch_b8
        const v8h w0 = *(const v8h*)(wrow);
        const v8h w1 = *(const v8h*)(wrow + 16);
        bF = __builtin_shufflevector(w0, w1, 0, 1, 2, 3, 4, 5, 6, 7,
                                             8, 9, 10, 11, 12, 13, 14, 15);
      } else {
        #pragma unroll
        for (int j = 0; j < 16; ++j) {
          const int k = kc * 32 + half * 8 + (j & 7) + (j >> 3) * 16;
          bF[j] = (k < KTOT && co < COUT) ? W[(size_t)co * KTOT + k] : (_Float16)0.f;
        }
      }
      acc[t] = __builtin_amdgcn_wmma_f32_16x16x32_f16(
          /*neg_a=*/false, aF, /*neg_b=*/false, bF,
          /*c_mod=*/(short)0, acc[t], /*reuse_a=*/false, /*reuse_b=*/false);
    }
  }

  // ---------------- store D: row = r + 8*half, col = lane&15 ----------------
  #pragma unroll
  for (int t = 0; t < COTILES; ++t) {
    const int co = t * 16 + m;
    if ((COUT % 16) == 0 || co < COUT) {
      const float bv = bias[co];
      #pragma unroll
      for (int r = 0; r < 8; ++r) {
        const int nd = nodeBase + half * 8 + r;
        if (nd < nNodes) {
          float v = acc[t][r] + bv;
          if (applyElu) v = (v > 0.f) ? v : (__expf(v) - 1.f);
          out[((size_t)b * nNodes + nd) * COUT + co] = (TOUT)v;
        }
      }
    }
  }
}

// ---------------------------------------------------------------------------
// Pool: rows = repeat(arange(nOut), 3) -> out row r owns entries r*3..r*3+2.
// f16 in/out, f32 accumulate.
// ---------------------------------------------------------------------------
__global__ void pool_k3(const _Float16* __restrict__ in, const int* __restrict__ cols,
                        const float* __restrict__ vals, _Float16* __restrict__ out,
                        int nOut, int nIn, int C)
{
  int i = blockIdx.x * blockDim.x + threadIdx.x;
  int total = 8 * nOut * C;
  if (i >= total) return;
  int c = i % C;
  int t = i / C;
  int r = t % nOut;
  int b = t / nOut;
  const _Float16* inb = in + (size_t)b * nIn * C;
  float s = 0.f;
  #pragma unroll
  for (int j = 0; j < 3; ++j) {
    int e = r * 3 + j;
    s += vals[e] * (float)inb[(size_t)cols[e] * C + c];
  }
  out[i] = (_Float16)s;
}

// ---------------------------------------------------------------------------
// enc linear: y[b,o] = sigmoid(dot(h[b,:40192], W[o,:]) + bias[o]); o in 0..127.
// Weight-streaming bound; one block per o, LDS tree reduce.
// ---------------------------------------------------------------------------
__global__ __launch_bounds__(256) void enc_linear(
    const _Float16* __restrict__ h, const float* __restrict__ W,
    const float* __restrict__ bias, float* __restrict__ z, float* __restrict__ mu)
{
  __shared__ float red[8][256];
  const int o = blockIdx.x;
  const int tid = threadIdx.x;
  const float* w = W + (size_t)o * 40192;
  float acc[8] = {};
  for (int k = tid; k < 40192; k += 256) {
    float wv = w[k];
    #pragma unroll
    for (int b = 0; b < 8; ++b) acc[b] += wv * (float)h[(size_t)b * 40192 + k];
  }
  #pragma unroll
  for (int b = 0; b < 8; ++b) red[b][tid] = acc[b];
  __syncthreads();
  for (int s = 128; s > 0; s >>= 1) {
    if (tid < s) {
      #pragma unroll
      for (int b = 0; b < 8; ++b) red[b][tid] += red[b][tid + s];
    }
    __syncthreads();
  }
  if (tid < 8) {
    float v = red[tid][0] + bias[o];
    v = 1.f / (1.f + __expf(-v));
    z[tid * 128 + o]  = v;
    mu[tid * 128 + o] = v;
  }
}

// ---------------------------------------------------------------------------
// dec linear: d[b,o] = dot(z[b,:128], W[o,:]) + bias[o]; o in 0..40191; f16 out.
// ---------------------------------------------------------------------------
__global__ void dec_linear(const float* __restrict__ z, const float* __restrict__ W,
                           const float* __restrict__ bias, _Float16* __restrict__ out)
{
  int o = blockIdx.x * blockDim.x + threadIdx.x;
  if (o >= 40192) return;
  const float4* w4 = (const float4*)(W + (size_t)o * 128);
  float bv = bias[o];
  float acc[8];
  #pragma unroll
  for (int b = 0; b < 8; ++b) acc[b] = bv;
  for (int k4 = 0; k4 < 32; ++k4) {
    float4 wv = w4[k4];
    #pragma unroll
    for (int b = 0; b < 8; ++b) {
      const float4 zv = *(const float4*)(z + b * 128 + k4 * 4);
      acc[b] += wv.x * zv.x + wv.y * zv.y + wv.z * zv.z + wv.w * zv.w;
    }
  }
  #pragma unroll
  for (int b = 0; b < 8; ++b) out[(size_t)b * 40192 + o] = (_Float16)acc[b];
}

// ---------------------------------------------------------------------------
extern "C" void kernel_launch(void* const* d_in, const int* in_sizes, int n_in,
                              void* d_out, int out_size, void* d_ws, size_t ws_size,
                              hipStream_t stream)
{
  (void)in_sizes; (void)n_in; (void)out_size; (void)ws_size;

  const float* x = (const float*)d_in[0];
  const int*   sp[4]    = {(const int*)d_in[1],  (const int*)d_in[2],
                           (const int*)d_in[3],  (const int*)d_in[4]};
  const int*   dcols[4] = {(const int*)d_in[9],  (const int*)d_in[10],
                           (const int*)d_in[11], (const int*)d_in[12]};
  const float* dvals[4] = {(const float*)d_in[13], (const float*)d_in[14],
                           (const float*)d_in[15], (const float*)d_in[16]};
  const int*   ucols[4] = {(const int*)d_in[21], (const int*)d_in[22],
                           (const int*)d_in[23], (const int*)d_in[24]};
  const float* uvals[4] = {(const float*)d_in[25], (const float*)d_in[26],
                           (const float*)d_in[27], (const float*)d_in[28]};
  const float* encW[4]  = {(const float*)d_in[29], (const float*)d_in[30],
                           (const float*)d_in[31], (const float*)d_in[32]};
  const float* encB[4]  = {(const float*)d_in[33], (const float*)d_in[34],
                           (const float*)d_in[35], (const float*)d_in[36]};
  const float* encLinW  = (const float*)d_in[37];
  const float* encLinB  = (const float*)d_in[38];
  const float* decLinW  = (const float*)d_in[39];
  const float* decLinB  = (const float*)d_in[40];
  const float* decW[4]  = {(const float*)d_in[41], (const float*)d_in[42],
                           (const float*)d_in[43], (const float*)d_in[44]};
  const float* decB[4]  = {(const float*)d_in[45], (const float*)d_in[46],
                           (const float*)d_in[47], (const float*)d_in[48]};
  const float* outW     = (const float*)d_in[49];
  const float* outB     = (const float*)d_in[50];

  const int N0 = 40000, N1 = 10000, N2 = 2500, N3 = 625, N4 = 157;

  // ---- workspace carve-up (all f16) ----
  _Float16* hA = (_Float16*)d_ws;                      // 20.48M halves
  _Float16* hB = hA + (size_t)8 * 40000 * 64;          // 20.48M halves
  _Float16* xh = hB + (size_t)8 * 40000 * 64;          // 0.96M halves
  _Float16* wp = xh + (size_t)8 * 40000 * 3;           // f16 weight copies
  const int wsz[9] = {32 * 27,   64 * 288,  128 * 576, 256 * 1152,  // enc 0..3
                      256 * 2304, 128 * 2304, 64 * 1152, 32 * 576,  // dec 0..3
                      3 * 288};                                     // head
  _Float16* wh[9];
  { _Float16* p = wp; for (int i = 0; i < 9; ++i) { wh[i] = p; p += wsz[i]; } }

  float* outp = (float*)d_out;                   // [8,40000,3]
  float* zp   = outp + 8 * 40000 * 3;            // [8,128]
  float* mup  = zp + 8 * 128;                    // [8,128]

  #define CVT(SRC, DST, N) \
    cvt_f32_f16<<<((N) + 255) / 256, 256, 0, stream>>>(SRC, DST, N)
  #define CONV(CI, CO, TO, IN, SPI, WW, BB, OUT, NN, ELU)                      \
    spiral_conv_wmma<CI, CO, TO>                                               \
        <<<dim3(((((NN) + 15) / 16) + 7) / 8, 8), 256, 0, stream>>>(           \
            IN, SPI, WW, BB, OUT, NN, ELU)
  #define POOL(IN, CO, VA, OUT, NOUT, NIN, C)                                  \
    pool_k3<<<(8*(NOUT)*(C)+255)/256, 256, 0, stream>>>(IN, CO, VA, OUT,       \
                                                        NOUT, NIN, C)

  // ---- one-shot f16 conversions (input + conv weights, ~2.4M elems) ----
  CVT(x, xh, 8 * 40000 * 3);
  CVT(encW[0], wh[0], wsz[0]);  CVT(encW[1], wh[1], wsz[1]);
  CVT(encW[2], wh[2], wsz[2]);  CVT(encW[3], wh[3], wsz[3]);
  CVT(decW[0], wh[4], wsz[4]);  CVT(decW[1], wh[5], wsz[5]);
  CVT(decW[2], wh[6], wsz[6]);  CVT(decW[3], wh[7], wsz[7]);
  CVT(outW,    wh[8], wsz[8]);

  // ---------------- encoder ----------------
  CONV(3,   32,  _Float16, xh, sp[0], wh[0], encB[0], hA, N0, 1);
  POOL(hA, dcols[0], dvals[0], hB, N1, N0, 32);
  CONV(32,  64,  _Float16, hB, sp[1], wh[1], encB[1], hA, N1, 1);
  POOL(hA, dcols[1], dvals[1], hB, N2, N1, 64);
  CONV(64,  128, _Float16, hB, sp[2], wh[2], encB[2], hA, N2, 1);
  POOL(hA, dcols[2], dvals[2], hB, N3, N2, 128);
  CONV(128, 256, _Float16, hB, sp[3], wh[3], encB[3], hA, N3, 1);
  POOL(hA, dcols[3], dvals[3], hB, N4, N3, 256);

  // ---------------- latent ----------------
  enc_linear<<<128, 256, 0, stream>>>(hB, encLinW, encLinB, zp, mup);
  dec_linear<<<(40192 + 255) / 256, 256, 0, stream>>>(zp, decLinW, decLinB, hA);

  // ---------------- decoder ----------------
  POOL(hA, ucols[3], uvals[3], hB, N3, N4, 256);
  CONV(256, 256, _Float16, hB, sp[3], wh[4], decB[0], hA, N3, 1);
  POOL(hA, ucols[2], uvals[2], hB, N2, N3, 256);
  CONV(256, 128, _Float16, hB, sp[2], wh[5], decB[1], hA, N2, 1);
  POOL(hA, ucols[1], uvals[1], hB, N1, N2, 128);
  CONV(128, 64,  _Float16, hB, sp[1], wh[6], decB[2], hA, N1, 1);
  POOL(hA, ucols[0], uvals[0], hB, N0, N1, 64);
  CONV(64,  32,  _Float16, hB, sp[0], wh[7], decB[3], hA, N0, 1);

  // output head (no ELU), f32 straight into d_out [8,40000,3]
  CONV(32,  3,   float, hA, sp[0], wh[8], outB, outp, N0, 0);

  #undef CVT
  #undef CONV
  #undef POOL
}
